// ScoringConnectivityGenerator_13211319402665
// MI455X (gfx1250) — matrix operand order repaired and checked by reading, not compiled, not run on hardware
//
#include <hip/hip_runtime.h>
#include <hip/hip_bf16.h>

typedef _Float16 half_t;
typedef _Float16 v16h __attribute__((ext_vector_type(16)));
typedef float    v8f  __attribute__((ext_vector_type(8)));

#define NNODE 19
#define NB    4096
#define EDG   342
#define ETOT  361           // 342 + 19 self loops
#define MROWS (NB * NNODE)  // 77824
#define MTILES (MROWS / 16) // 4864

// ---------------------------------------------------------------------------
// Fragment-layout packing helpers (CDNA5 ISA 7.12.2, wave32)
//   A (16x32 f16): lane l (l<16): row=l, K = e + (e&8)        (e = 0..15)
//                  lane l (>=16): row=l-16, K = 8 + e + (e&8)
//   B (32x16 f16): lane l: col = l&15, K = 8*(l>>4) + e + (e&8)
//   C/D (16x16 f32): vgpr f: row = f + 8*(l>>4), col = l&15
// Packed linear index: ((tile*ktiles + kt)*32 + lane)*16 + e
// ---------------------------------------------------------------------------

// Pack weight matrix W[K x N] (f32 row-major) into B-fragment order (f16).
__global__ void pack_w_kernel(const float* __restrict__ W, half_t* __restrict__ out,
                              int K, int N) {
    int idx = blockIdx.x * blockDim.x + threadIdx.x;
    if (idx >= K * N) return;
    int e    = idx & 15;
    int lane = (idx >> 4) & 31;
    int t    = idx >> 9;
    int ktiles = K >> 5;
    int kt = t % ktiles;
    int nt = t / ktiles;
    int k = kt * 32 + 8 * (lane >> 4) + e + (e & 8);
    int n = nt * 16 + (lane & 15);
    out[idx] = (half_t)W[(size_t)k * N + n];
}

// Pack x[MROWS x 256] (f32 row-major) into A-fragment order (f16).
__global__ void pack_x_kernel(const float* __restrict__ X, half_t* __restrict__ out) {
    int idx = blockIdx.x * blockDim.x + threadIdx.x;   // < MROWS*256
    int e    = idx & 15;
    int lane = (idx >> 4) & 31;
    int t    = idx >> 9;
    int kt = t & 7;          // 256/32 = 8 k-tiles
    int mt = t >> 3;
    int m = mt * 16 + (lane & 15);
    int k = kt * 32 + 8 * (lane >> 4) + e + (e & 8);
    out[idx] = (half_t)X[(size_t)m * 256 + k];
}

// ---------------------------------------------------------------------------
// GEMM: H[MROWS x Ncol] (f32) = Apack[MROWS x K] * Bpack[K x Ncol]
// block = 256 (8 waves). wave -> one 16-row M tile x four 16-col N tiles.
// grid = (MTILES/8, Ncol/64)
// ---------------------------------------------------------------------------
__global__ __launch_bounds__(256) void gemm_wmma_kernel(
    const half_t* __restrict__ A, const half_t* __restrict__ B,
    float* __restrict__ H, int ktiles, int Ncol) {
    const int lane = threadIdx.x & 31;
    const int wid  = threadIdx.x >> 5;
    const int m_tile = blockIdx.x * 8 + wid;
    const int n0 = blockIdx.y * 4;                    // first of 4 n-tiles

    const size_t nstride = (size_t)ktiles * 512;      // halfs per n-tile column block
    const half_t* ab = A + (size_t)m_tile * nstride + (size_t)lane * 16;
    const half_t* bb = B + (size_t)n0 * nstride + (size_t)lane * 16;

    v8f acc0 = {}, acc1 = {}, acc2 = {}, acc3 = {};
    for (int kt = 0; kt < ktiles; ++kt) {
        const size_t ko = (size_t)kt * 512;
        v16h a  = *(const v16h*)(ab + ko);
        v16h b0 = *(const v16h*)(bb + ko);
        v16h b1 = *(const v16h*)(bb + ko + nstride);
        v16h b2 = *(const v16h*)(bb + ko + 2 * nstride);
        v16h b3 = *(const v16h*)(bb + ko + 3 * nstride);
        acc0 = __builtin_amdgcn_wmma_f32_16x16x32_f16(false, a, false, b0, (short)0, acc0, false, false);
        acc1 = __builtin_amdgcn_wmma_f32_16x16x32_f16(false, a, false, b1, (short)0, acc1, false, false);
        acc2 = __builtin_amdgcn_wmma_f32_16x16x32_f16(false, a, false, b2, (short)0, acc2, false, false);
        acc3 = __builtin_amdgcn_wmma_f32_16x16x32_f16(false, a, false, b3, (short)0, acc3, false, false);
    }

    const int g = lane >> 4, col = lane & 15;
#pragma unroll
    for (int f = 0; f < 8; ++f) {
        size_t row = (size_t)m_tile * 16 + f + 8 * g;
        float* hp = H + row * Ncol + (size_t)n0 * 16 + col;
        hp[0]  = acc0[f];
        hp[16] = acc1[f];
        hp[32] = acc2[f];
        hp[48] = acc3[f];
    }
}

// ---------------------------------------------------------------------------
// Attention logits: als[row] = h[row]·a_src, ald[row] = h[row]·a_dst
// One wave per row; grid = MROWS/8, block = 256.
// ---------------------------------------------------------------------------
__global__ __launch_bounds__(256) void logits_kernel(
    const float* __restrict__ H, const float* __restrict__ a_s,
    const float* __restrict__ a_d, float* __restrict__ als,
    float* __restrict__ ald, int C) {
    const int lane = threadIdx.x & 31;
    const int wid  = threadIdx.x >> 5;
    const size_t row = (size_t)blockIdx.x * 8 + wid;
    const float* h = H + row * C;
    float s = 0.f, d = 0.f;
    for (int c = lane; c < C; c += 32) {
        float v = h[c];
        s += v * a_s[c];
        d += v * a_d[c];
    }
#pragma unroll
    for (int off = 16; off; off >>= 1) {
        s += __shfl_xor(s, off, 32);
        d += __shfl_xor(d, off, 32);
    }
    if (lane == 0) { als[row] = s; ald[row] = d; }
}

// ---------------------------------------------------------------------------
// Per-graph attention softmax + aggregation + bias (+ReLU).
// Writes f16 output in A-fragment layout (for next GEMM) and/or f32 output.
// grid = NB, block = 256.
// ---------------------------------------------------------------------------
#define CHUNK 128
__global__ __launch_bounds__(256) void aggregate_kernel(
    const float* __restrict__ H, const float* __restrict__ als,
    const float* __restrict__ ald, const int* __restrict__ ei,
    const float* __restrict__ bias, half_t* __restrict__ apackOut,
    float* __restrict__ f32out, int C, int doRelu, int doF16, int doF32) {
    const int b   = blockIdx.x;
    const int tid = threadIdx.x;

    __shared__ float s_as[NNODE], s_ad[NNODE];
    __shared__ float elog[ETOT];
    __shared__ int   es[ETOT], ed[ETOT];
    __shared__ float Adj[NNODE][NNODE + 1];
    __shared__ float hch[NNODE][CHUNK + 1];

    if (tid < NNODE) {
        s_as[tid] = als[b * NNODE + tid];
        s_ad[tid] = ald[b * NNODE + tid];
    }
    for (int i = tid; i < ETOT; i += 256) {
        es[i] = (i < EDG) ? ei[i]       : (i - EDG);
        ed[i] = (i < EDG) ? ei[EDG + i] : (i - EDG);
    }
    for (int i = tid; i < NNODE * (NNODE + 1); i += 256)
        ((float*)Adj)[i] = 0.f;
    __syncthreads();
    for (int i = tid; i < ETOT; i += 256) {
        float v = s_as[es[i]] + s_ad[ed[i]];
        elog[i] = v > 0.f ? v : 0.2f * v;          // leaky_relu(0.2)
    }
    __syncthreads();
    if (tid < NNODE) {                              // thread owns one dst node
        const int d = tid;
        float m = -1e30f;
        for (int i = 0; i < ETOT; ++i) if (ed[i] == d) m = fmaxf(m, elog[i]);
        float s = 0.f;
        for (int i = 0; i < ETOT; ++i) if (ed[i] == d) s += __expf(elog[i] - m);
        float inv = 1.f / s;
        for (int i = 0; i < ETOT; ++i)
            if (ed[i] == d) Adj[d][es[i]] += __expf(elog[i] - m) * inv;
    }
    __syncthreads();

    const int ktiles = C >> 5;
    for (int c0 = 0; c0 < C; c0 += CHUNK) {
        for (int i = tid; i < NNODE * CHUNK; i += 256) {
            int n = i / CHUNK, cc = i % CHUNK;
            hch[n][cc] = H[((size_t)(b * NNODE + n)) * C + c0 + cc];
        }
        __syncthreads();
        for (int i = tid; i < NNODE * CHUNK; i += 256) {
            int n = i / CHUNK, cc = i % CHUNK;
            int c = c0 + cc;
            float acc = bias[c];
#pragma unroll
            for (int m = 0; m < NNODE; ++m) acc += Adj[n][m] * hch[m][cc];
            if (doRelu) acc = fmaxf(acc, 0.f);
            const int row = b * NNODE + n;
            if (doF32) f32out[(size_t)row * C + c] = acc;
            if (doF16) {
                // invert A-fragment mapping: k = 8g + e + (e&8)
                int mt = row >> 4, lm = row & 15;
                int kt = c >> 5,  kk = c & 31;
                int hi = kk >> 4;
                int g  = (kk >> 3) & 1;
                int e  = (kk & 7) + 8 * hi;
                size_t idx = (((size_t)mt * ktiles + kt) * 32 + (g * 16 + lm)) * 16 + e;
                apackOut[idx] = (half_t)acc;
            }
        }
        __syncthreads();
    }
}

// ---------------------------------------------------------------------------
// Scoring: score[b,n] = sigmoid( h4[b,n,:] · (sum_m ws[m]*h4[b,m,:]) + bs )
// grid = NB, block = 256.
// ---------------------------------------------------------------------------
__global__ __launch_bounds__(256) void score_kernel(
    const float* __restrict__ h4, const float* __restrict__ wsv,
    const float* __restrict__ bs, float* __restrict__ out) {
    const int b = blockIdx.x, tid = threadIdx.x;
    __shared__ float t[256];
    const float* hb = h4 + (size_t)b * NNODE * 256;
    float acc = 0.f;
#pragma unroll
    for (int m = 0; m < NNODE; ++m) acc += wsv[m] * hb[m * 256 + tid];
    t[tid] = acc;
    __syncthreads();
    if (tid < NNODE) {
        float s = 0.f;
        for (int c = 0; c < 256; ++c) s += hb[tid * 256 + c] * t[c];
        out[b * NNODE + tid] = 1.f / (1.f + __expf(-(s + bs[0])));
    }
}

// ---------------------------------------------------------------------------
extern "C" void kernel_launch(void* const* d_in, const int* in_sizes, int n_in,
                              void* d_out, int out_size, void* d_ws, size_t ws_size,
                              hipStream_t stream) {
    const float* x   = (const float*)d_in[0];
    const int*   ei  = (const int*)d_in[1];
    const float* W1  = (const float*)d_in[3];
    const float* as1 = (const float*)d_in[4];
    const float* ad1 = (const float*)d_in[5];
    const float* b1  = (const float*)d_in[6];
    const float* W2  = (const float*)d_in[7];
    const float* as2 = (const float*)d_in[8];
    const float* ad2 = (const float*)d_in[9];
    const float* b2  = (const float*)d_in[10];
    const float* W3  = (const float*)d_in[11];
    const float* as3 = (const float*)d_in[12];
    const float* ad3 = (const float*)d_in[13];
    const float* b3  = (const float*)d_in[14];
    const float* W4  = (const float*)d_in[15];
    const float* as4 = (const float*)d_in[16];
    const float* ad4 = (const float*)d_in[17];
    const float* b4  = (const float*)d_in[18];
    const float* wsv = (const float*)d_in[19];
    const float* bsv = (const float*)d_in[20];
    float* out = (float*)d_out;

    // Workspace carve-up
    char* p = (char*)d_ws;
    half_t* wp1 = (half_t*)p; p += (size_t)256 * 512 * 2;
    half_t* wp2 = (half_t*)p; p += (size_t)512 * 512 * 2;
    half_t* wp3 = (half_t*)p; p += (size_t)512 * 512 * 2;
    half_t* wp4 = (half_t*)p; p += (size_t)512 * 256 * 2;
    half_t* apack = (half_t*)p; p += (size_t)MROWS * 512 * 2;
    float*  Hbuf  = (float*)p;  p += (size_t)MROWS * 512 * 4;
    float*  h4    = (float*)p;  p += (size_t)MROWS * 256 * 4;
    float*  als   = (float*)p;  p += (size_t)MROWS * 4;
    float*  ald   = (float*)p;  p += (size_t)MROWS * 4;
    (void)ws_size; (void)n_in; (void)in_sizes; (void)out_size;

    // Pack weights + input into WMMA fragment layouts
    pack_w_kernel<<<(256 * 512 + 255) / 256, 256, 0, stream>>>(W1, wp1, 256, 512);
    pack_w_kernel<<<(512 * 512 + 255) / 256, 256, 0, stream>>>(W2, wp2, 512, 512);
    pack_w_kernel<<<(512 * 512 + 255) / 256, 256, 0, stream>>>(W3, wp3, 512, 512);
    pack_w_kernel<<<(512 * 256 + 255) / 256, 256, 0, stream>>>(W4, wp4, 512, 256);
    pack_x_kernel<<<(MROWS * 256) / 256, 256, 0, stream>>>(x, apack);

    dim3 blk(256);

    // Layer 1: K=256 -> C=512, ReLU
    gemm_wmma_kernel<<<dim3(MTILES / 8, 512 / 64), blk, 0, stream>>>(apack, wp1, Hbuf, 256 / 32, 512);
    logits_kernel<<<MROWS / 8, blk, 0, stream>>>(Hbuf, as1, ad1, als, ald, 512);
    aggregate_kernel<<<NB, blk, 0, stream>>>(Hbuf, als, ald, ei, b1, apack, nullptr, 512, 1, 1, 0);

    // Layer 2: K=512 -> C=512, ReLU
    gemm_wmma_kernel<<<dim3(MTILES / 8, 512 / 64), blk, 0, stream>>>(apack, wp2, Hbuf, 512 / 32, 512);
    logits_kernel<<<MROWS / 8, blk, 0, stream>>>(Hbuf, as2, ad2, als, ald, 512);
    aggregate_kernel<<<NB, blk, 0, stream>>>(Hbuf, als, ald, ei, b2, apack, nullptr, 512, 1, 1, 0);

    // Layer 3: K=512 -> C=512, ReLU
    gemm_wmma_kernel<<<dim3(MTILES / 8, 512 / 64), blk, 0, stream>>>(apack, wp3, Hbuf, 512 / 32, 512);
    logits_kernel<<<MROWS / 8, blk, 0, stream>>>(Hbuf, as3, ad3, als, ald, 512);
    aggregate_kernel<<<NB, blk, 0, stream>>>(Hbuf, als, ald, ei, b3, apack, nullptr, 512, 1, 1, 0);

    // Layer 4: K=512 -> C=256, no activation, keep f32 for scoring
    gemm_wmma_kernel<<<dim3(MTILES / 8, 256 / 64), blk, 0, stream>>>(apack, wp4, Hbuf, 512 / 32, 256);
    logits_kernel<<<MROWS / 8, blk, 0, stream>>>(Hbuf, as4, ad4, als, ald, 256);
    aggregate_kernel<<<NB, blk, 0, stream>>>(Hbuf, als, ald, ei, b4, nullptr, h4, 256, 0, 0, 1);

    // conn = h h^T ; scores = sigmoid(conn @ ws + bs)
    score_kernel<<<NB, blk, 0, stream>>>(h4, wsv, bsv, out);
}
